// CaptionDecoder_16277926052182
// MI455X (gfx1250) — matrix operand order repaired and checked by reading, not compiled, not run on hardware
//
#include <hip/hip_runtime.h>
#include <hip/hip_bf16.h>
#include <math.h>

// ---------------------------------------------------------------------------
// CaptionDecoder on MI455X (gfx1250): 2-layer LSTM + FC head + greedy/teacher
// feedback, 64 sequential steps. All GEMMs via v_wmma_f32_16x16x32_bf16.
// A-operands staged in LDS (ds_load_b128) to avoid VGPR spills in K loops.
// ---------------------------------------------------------------------------

#define HID       512
#define EMB       512
#define BATCH     32
#define TSTEPS    64
#define VOCAB     30522
#define VOCABPAD  30528             // padded to multiple of 16 for WMMA tiles
#define NTILES_FC (VOCABPAD / 16)   // 1908
#define START_TOKEN 101

typedef __attribute__((ext_vector_type(16))) __bf16 v16bf;
typedef __attribute__((ext_vector_type(8)))  float  v8f;

union Frag {
    v16bf v;
    unsigned short s[16];
    uint4 q[2];
};

__device__ __forceinline__ unsigned short f2bf(float f) {
    union { float f; unsigned u; } x; x.f = f;
    unsigned r = x.u + 0x7FFFu + ((x.u >> 16) & 1u);   // round-to-nearest-even
    return (unsigned short)(r >> 16);
}

__device__ __forceinline__ float sigmoidf_(float x) {
    return 1.0f / (1.0f + __expf(-x));
}

// A fragment from an LDS-resident row-major bf16 matrix [rows][ldk].
// Per ISA A-layout: lanes 0-15 hold K = kb+[0..7, 16..23] of row (lane&15),
// lanes 16-31 hold K = kb+[8..15, 24..31].
__device__ __forceinline__ v16bf loadA_lds(const unsigned short* Abf, int ldk,
                                           int mbase, int kb, int lane) {
    Frag f;
    int row = mbase + (lane & 15);
    const unsigned short* p = Abf + row * ldk + kb + ((lane >> 4) << 3);
    f.q[0] = *reinterpret_cast<const uint4*>(p);
    f.q[1] = *reinterpret_cast<const uint4*>(p + 16);
    return f.v;
}

// B fragment: B[K,N] = W[N,K], W row-major [Nout][ldk] bf16 in global memory.
// Lane holds column N = nbase+(lane&15), K = kb + (lane>>4)*8 + [0..7, 16..23].
__device__ __forceinline__ v16bf loadB(const unsigned short* __restrict__ Wbf,
                                       int ldk, int nbase, int kb, int lane) {
    Frag f;
    int col = nbase + (lane & 15);
    const unsigned short* p = Wbf + (size_t)col * ldk + kb + ((lane >> 4) << 3);
    f.q[0] = *reinterpret_cast<const uint4*>(p);
    f.q[1] = *reinterpret_cast<const uint4*>(p + 16);
    return f.v;
}

__device__ __forceinline__ v8f wmma_bf16(v16bf a, v16bf b, v8f c) {
    return __builtin_amdgcn_wmma_f32_16x16x32_bf16(
        /*neg_a=*/false, a, /*neg_b=*/false, b,
        /*c_mod=*/(short)0, c, /*reuse_a=*/false, /*reuse_b=*/false);
}

// ------------------------- one-time setup kernels --------------------------

__global__ void cvt_bf16_kernel(const float* __restrict__ src,
                                unsigned short* __restrict__ dst, int n) {
    int i = blockIdx.x * blockDim.x + threadIdx.x;
    if (i < n) dst[i] = f2bf(src[i]);
}

__global__ void cvt_fcw_kernel(const float* __restrict__ src,
                               unsigned short* __restrict__ dst) {
    int i = blockIdx.x * blockDim.x + threadIdx.x;
    int n = VOCABPAD * HID;
    if (i < n) dst[i] = (i < VOCAB * HID) ? f2bf(src[i]) : (unsigned short)0;
}

__global__ void init_state_kernel(const float* __restrict__ fused,
                                  unsigned short* __restrict__ h0bf,
                                  unsigned short* __restrict__ h1bf,
                                  float* __restrict__ c0,
                                  float* __restrict__ c1,
                                  int* __restrict__ tok) {
    int i = blockIdx.x * blockDim.x + threadIdx.x;
    if (i < BATCH * HID) {
        unsigned short h = f2bf(fused[i]);
        h0bf[i] = h; h1bf[i] = h;
        c0[i] = 0.0f; c1[i] = 0.0f;
    }
    if (i < BATCH) tok[i] = START_TOKEN;
}

// ------------------------------ LSTM step ----------------------------------
// Single workgroup, 512 threads = 16 waves.
// Gate GEMM: out [32 x 2048] = As @ Wih^T + Ah @ Whh^T; As/Ah live in LDS.

__device__ __forceinline__ void gemm_gates(const unsigned short* As,   // LDS [32][512]
                                           const unsigned short* Ah,   // LDS [32][512]
                                           const unsigned short* __restrict__ Wih,
                                           const unsigned short* __restrict__ Whh,
                                           float* __restrict__ gates,
                                           int wave, int lane) {
    // 2 M-tiles x 128 N-tiles = 256 tiles over 16 waves
    for (int tile = wave; tile < 256; tile += 16) {
        int mt = tile & 1, nt = tile >> 1;
        int mbase = mt * 16, nbase = nt * 16;
        v8f acc = {0.f, 0.f, 0.f, 0.f, 0.f, 0.f, 0.f, 0.f};
        #pragma unroll 2
        for (int kb = 0; kb < HID; kb += 32) {
            v16bf a0 = loadA_lds(As, HID, mbase, kb, lane);
            v16bf b0 = loadB(Wih, HID, nbase, kb, lane);
            acc = wmma_bf16(a0, b0, acc);
            v16bf a1 = loadA_lds(Ah, HID, mbase, kb, lane);
            v16bf b1 = loadB(Whh, HID, nbase, kb, lane);
            acc = wmma_bf16(a1, b1, acc);
        }
        int col   = nbase + (lane & 15);
        int rbase = mbase + ((lane >> 4) << 3);
        #pragma unroll
        for (int r = 0; r < 8; ++r)
            gates[(size_t)(rbase + r) * (4 * HID) + col] = acc[r];
    }
}

__global__ void __launch_bounds__(512)
lstm_step_kernel(const float* __restrict__ embedding,
                 const unsigned short* __restrict__ wih0,
                 const unsigned short* __restrict__ whh0,
                 const float* __restrict__ bih0, const float* __restrict__ bhh0,
                 const unsigned short* __restrict__ wih1,
                 const unsigned short* __restrict__ whh1,
                 const float* __restrict__ bih1, const float* __restrict__ bhh1,
                 unsigned short* __restrict__ h0bf,
                 unsigned short* __restrict__ h1bf,
                 float* __restrict__ c0, float* __restrict__ c1,
                 const int* __restrict__ tok,
                 float* __restrict__ gates) {
    __shared__ unsigned short As[BATCH * HID];   // 32 KB: x, then h0_new
    __shared__ unsigned short Ah[BATCH * HID];   // 32 KB: h0_old, then h1_old

    int tid  = threadIdx.x;
    int lane = tid & 31;
    int wave = tid >> 5;

    // phase 0: As <- bf16(embedding[tok]);  Ah <- h0_old (vectorized)
    for (int i = tid; i < BATCH * EMB; i += 512) {
        int b = i >> 9, e = i & (EMB - 1);
        As[i] = f2bf(embedding[(size_t)tok[b] * EMB + e]);
    }
    {
        const uint4* src = reinterpret_cast<const uint4*>(h0bf);
        uint4* dst = reinterpret_cast<uint4*>(Ah);
        for (int i = tid; i < BATCH * HID / 8; i += 512) dst[i] = src[i];
    }
    __syncthreads();

    // phase 1: layer-0 gate GEMM
    gemm_gates(As, Ah, wih0, whh0, gates, wave, lane);
    __syncthreads();

    // phase 2: layer-0 elementwise; As <- h0_new (also to global state);
    //          Ah <- h1_old for layer 1
    for (int i = tid; i < BATCH * HID; i += 512) {
        int b = i >> 9, j = i & (HID - 1);
        const float* g = gates + (size_t)b * (4 * HID);
        float gi = g[j]           + bih0[j]           + bhh0[j];
        float gf = g[HID + j]     + bih0[HID + j]     + bhh0[HID + j];
        float gg = g[2 * HID + j] + bih0[2 * HID + j] + bhh0[2 * HID + j];
        float go = g[3 * HID + j] + bih0[3 * HID + j] + bhh0[3 * HID + j];
        float cn = sigmoidf_(gf) * c0[i] + sigmoidf_(gi) * tanhf(gg);
        c0[i] = cn;
        unsigned short hn = f2bf(sigmoidf_(go) * tanhf(cn));
        As[i] = hn;
        h0bf[i] = hn;
        Ah[i] = h1bf[i];
    }
    __syncthreads();

    // phase 3: layer-1 gate GEMM
    gemm_gates(As, Ah, wih1, whh1, gates, wave, lane);
    __syncthreads();

    // phase 4: layer-1 elementwise; h1_new -> global state (FC input)
    for (int i = tid; i < BATCH * HID; i += 512) {
        int b = i >> 9, j = i & (HID - 1);
        const float* g = gates + (size_t)b * (4 * HID);
        float gi = g[j]           + bih1[j]           + bhh1[j];
        float gf = g[HID + j]     + bih1[HID + j]     + bhh1[HID + j];
        float gg = g[2 * HID + j] + bih1[2 * HID + j] + bhh1[2 * HID + j];
        float go = g[3 * HID + j] + bih1[3 * HID + j] + bhh1[3 * HID + j];
        float cn = sigmoidf_(gf) * c1[i] + sigmoidf_(gi) * tanhf(gg);
        c1[i] = cn;
        h1bf[i] = f2bf(sigmoidf_(go) * tanhf(cn));
    }
}

// ------------------------------ FC head ------------------------------------
// logits[b, v] = h1 @ fc_W^T + fc_b. h1 staged in LDS per block; one 16-wide
// N-tile per wave, both 16-row M-tiles per wave, K=512 in chunks of 32.

__global__ void __launch_bounds__(256)
fc_kernel(const unsigned short* __restrict__ h1bf,
          const unsigned short* __restrict__ fcw,
          const float* __restrict__ fcb,
          float* __restrict__ out, int t) {
    __shared__ unsigned short Alds[BATCH * HID];  // 32 KB

    int tid  = threadIdx.x;
    int lane = tid & 31;
    int wid  = tid >> 5;

    // cooperative fill of A (h1, 32x512 bf16)
    {
        const uint4* src = reinterpret_cast<const uint4*>(h1bf);
        uint4* dst = reinterpret_cast<uint4*>(Alds);
        for (int i = tid; i < BATCH * HID / 8; i += 256) dst[i] = src[i];
    }
    __syncthreads();

    int nt = blockIdx.x * 8 + wid;          // wave-uniform
    if (nt < NTILES_FC) {                   // uniform branch: EXEC stays full
        int nbase = nt * 16;

        v8f acc0 = {0.f, 0.f, 0.f, 0.f, 0.f, 0.f, 0.f, 0.f};
        v8f acc1 = {0.f, 0.f, 0.f, 0.f, 0.f, 0.f, 0.f, 0.f};
        #pragma unroll 4
        for (int kb = 0; kb < HID; kb += 32) {
            v16bf bfr = loadB(fcw, HID, nbase, kb, lane);
            v16bf a0  = loadA_lds(Alds, HID, 0,  kb, lane);
            v16bf a1  = loadA_lds(Alds, HID, 16, kb, lane);
            acc0 = wmma_bf16(a0, bfr, acc0);
            acc1 = wmma_bf16(a1, bfr, acc1);
        }

        int col = nbase + (lane & 15);
        if (col < VOCAB) {
            float bias = fcb[col];
            int rb = (lane >> 4) << 3;
            #pragma unroll
            for (int r = 0; r < 8; ++r) {
                int b0 = rb + r;
                int b1 = 16 + rb + r;
                out[(size_t)b0 * TSTEPS * VOCAB + (size_t)t * VOCAB + col] = acc0[r] + bias;
                out[(size_t)b1 * TSTEPS * VOCAB + (size_t)t * VOCAB + col] = acc1[r] + bias;
            }
        }
    }
}

// --------------------- argmax + teacher-forcing select ---------------------

__global__ void __launch_bounds__(256)
argmax_select_kernel(const float* __restrict__ out,
                     const int* __restrict__ target_captions,
                     const int* __restrict__ teacher_mask,
                     int* __restrict__ tok, int t) {
    __shared__ float sval[256];
    __shared__ int   sidx[256];
    int b = blockIdx.x;
    int tid = threadIdx.x;
    const float* row = out + (size_t)b * TSTEPS * VOCAB + (size_t)t * VOCAB;

    float best = -3.4e38f;
    int bidx = 0;
    for (int v = tid; v < VOCAB; v += 256) {
        float x = row[v];
        if (x > best) { best = x; bidx = v; }   // increasing v -> first max kept
    }
    sval[tid] = best; sidx[tid] = bidx;
    __syncthreads();
    for (int s = 128; s > 0; s >>= 1) {
        if (tid < s) {
            float ov = sval[tid + s]; int oi = sidx[tid + s];
            if (ov > sval[tid] || (ov == sval[tid] && oi < sidx[tid])) {
                sval[tid] = ov; sidx[tid] = oi;
            }
        }
        __syncthreads();
    }
    if (tid == 0) {
        bool teach = (teacher_mask[t] != 0) && (t < TSTEPS - 1);
        int nxt = teach ? target_captions[b * TSTEPS + (t + 1)] : sidx[0];
        tok[b] = nxt;
    }
}

// ------------------------------ launch -------------------------------------

extern "C" void kernel_launch(void* const* d_in, const int* in_sizes, int n_in,
                              void* d_out, int out_size, void* d_ws, size_t ws_size,
                              hipStream_t stream) {
    const float* fused = (const float*)d_in[0];
    const int*   tgt   = (const int*)d_in[1];
    const int*   tmask = (const int*)d_in[2];
    const float* emb   = (const float*)d_in[3];
    const float* Wih0  = (const float*)d_in[4];
    const float* Whh0  = (const float*)d_in[5];
    const float* bih0  = (const float*)d_in[6];
    const float* bhh0  = (const float*)d_in[7];
    const float* Wih1  = (const float*)d_in[8];
    const float* Whh1  = (const float*)d_in[9];
    const float* bih1  = (const float*)d_in[10];
    const float* bhh1  = (const float*)d_in[11];
    const float* fcW   = (const float*)d_in[12];
    const float* fcb   = (const float*)d_in[13];
    float* out = (float*)d_out;

    // workspace carve-out (~40 MB)
    char* p = (char*)d_ws;
    auto carve = [&](size_t bytes) {
        void* r = (void*)p;
        p += (bytes + 255) & ~(size_t)255;
        return r;
    };
    unsigned short* fcWbf  = (unsigned short*)carve((size_t)VOCABPAD * HID * 2);
    unsigned short* wih0bf = (unsigned short*)carve((size_t)4 * HID * HID * 2);
    unsigned short* whh0bf = (unsigned short*)carve((size_t)4 * HID * HID * 2);
    unsigned short* wih1bf = (unsigned short*)carve((size_t)4 * HID * HID * 2);
    unsigned short* whh1bf = (unsigned short*)carve((size_t)4 * HID * HID * 2);
    float*          gates  = (float*)carve((size_t)BATCH * 4 * HID * 4);
    unsigned short* h0bf   = (unsigned short*)carve((size_t)BATCH * HID * 2);
    unsigned short* h1bf   = (unsigned short*)carve((size_t)BATCH * HID * 2);
    float*          c0     = (float*)carve((size_t)BATCH * HID * 4);
    float*          c1     = (float*)carve((size_t)BATCH * HID * 4);
    int*            tok    = (int*)carve(BATCH * sizeof(int));

    // one-time conversions (re-run every call: deterministic)
    {
        int n = 4 * HID * HID;
        int blocks = (n + 255) / 256;
        cvt_bf16_kernel<<<blocks, 256, 0, stream>>>(Wih0, wih0bf, n);
        cvt_bf16_kernel<<<blocks, 256, 0, stream>>>(Whh0, whh0bf, n);
        cvt_bf16_kernel<<<blocks, 256, 0, stream>>>(Wih1, wih1bf, n);
        cvt_bf16_kernel<<<blocks, 256, 0, stream>>>(Whh1, whh1bf, n);
        int nf = VOCABPAD * HID;
        cvt_fcw_kernel<<<(nf + 255) / 256, 256, 0, stream>>>(fcW, fcWbf);
        init_state_kernel<<<(BATCH * HID + 255) / 256, 256, 0, stream>>>(
            fused, h0bf, h1bf, c0, c1, tok);
    }

    int fc_blocks = (NTILES_FC + 7) / 8;   // 8 waves per block
    for (int t = 0; t < TSTEPS; ++t) {
        lstm_step_kernel<<<1, 512, 0, stream>>>(
            emb, wih0bf, whh0bf, bih0, bhh0,
            wih1bf, whh1bf, bih1, bhh1,
            h0bf, h1bf, c0, c1, tok, gates);
        fc_kernel<<<fc_blocks, 256, 0, stream>>>(h1bf, fcWbf, fcb, out, t);
        if (t < TSTEPS - 1) {
            argmax_select_kernel<<<BATCH, 256, 0, stream>>>(out, tgt, tmask, tok, t);
        }
    }
}
